// MultiheadAttention_66159676227890
// MI455X (gfx1250) — matrix operand order repaired and checked by reading, not compiled, not run on hardware
//
#include <hip/hip_runtime.h>
#include <hip/hip_bf16.h>

// ---------------------------------------------------------------------------
// MHA forward for MI455X (gfx1250, wave32).
// Roofline: 402 MB attention store => >=17us at 23.3 TB/s.  bf16 WMMA
// 16x16x32 (16384 FLOP/instr, f32 accumulate) keeps the 64.5 GFLOP of GEMM
// work (~4M WMMA issues) under the HBM store floor, unlike fp32 WMMA.
// CDNA5 paths used: v_wmma_f32_16x16x32_bf16, global_load_async_to_lds_b128
// (+ s_wait_asynccnt), global_prefetch_b8, split-barrier workgroup sync.
// ---------------------------------------------------------------------------

typedef float  v8f   __attribute__((ext_vector_type(8)));
typedef __bf16 v8bf  __attribute__((ext_vector_type(8)));
typedef __bf16 v16bf __attribute__((ext_vector_type(16)));

#define EMBED   768
#define QKV_N   2304
#define HEADS   12
#define HDIM    64
#define BATCH   8
#define SEQ     1024
#define ROWS    (BATCH * SEQ)          // 8192

__device__ __forceinline__ unsigned short f2bf(float f) {
  union { float f; unsigned int u; } v; v.f = f;
  const unsigned int r = v.u + 0x7FFFu + ((v.u >> 16) & 1u);   // RNE
  return (unsigned short)(r >> 16);
}

// Packed f32x2 -> bf16x2 (v_cvt_pk_bf16_f32 when the toolchain has it).
__device__ __forceinline__ unsigned cvt_pk_bf16(float lo, float hi) {
#if __has_builtin(__builtin_amdgcn_cvt_pk_bf16_f32)
  auto r = __builtin_amdgcn_cvt_pk_bf16_f32(lo, hi);
  unsigned u; __builtin_memcpy(&u, &r, 4); return u;
#else
  return (unsigned)f2bf(lo) | ((unsigned)f2bf(hi) << 16);
#endif
}

__device__ __forceinline__ v16bf cat8(v8bf lo, v8bf hi) {
  return __builtin_shufflevector(lo, hi, 0,1,2,3,4,5,6,7,8,9,10,11,12,13,14,15);
}

__device__ __forceinline__ v8f wmma_bf16(v16bf a, v16bf b, v8f c) {
  // D(16x16 f32) = A(16x32 bf16) * B(32x16 bf16) + C
  return __builtin_amdgcn_wmma_f32_16x16x32_bf16(
      /*neg_a=*/false, a, /*neg_b=*/false, b,
      /*c_mod=*/(short)0, c, /*reuse_a=*/false, /*reuse_b=*/false);
}

// Per-lane bf16 A fragment (16x32, row m = lane%16, half h = lane/16):
//   K-run  [16h, 16h+16)                    -> two contiguous b128 loads
// Per-lane bf16 B fragment (32x16, col n = lane%16):
//   K-run  [16h, 16h+16)                    -> two contiguous b128 loads

// ---------------------------------------------------------------------------
// C[M,N] = A[M,K=768] @ W[K,N] + bias[N].
// A_BF16: A already bf16 (values path, staged with ASYNC-to-LDS copies).
// OUT_BF16: emit bf16 (qkv path).
// Block 256 = 8 waves (2x4), block tile 32x128, K chunk 64 (4 WMMA/wave).
// ---------------------------------------------------------------------------
template <bool A_BF16, bool OUT_BF16>
__global__ __launch_bounds__(256) void gemm_bias_kernel(
    const void* __restrict__ Av, const float* __restrict__ Wm,
    const float* __restrict__ bias, void* __restrict__ Cv, int N)
{
  constexpr int K = EMBED;
  __shared__ unsigned short As[32][72];    // row-major A panel, 144B stride
  __shared__ unsigned short Bt[128][72];   // TRANSPOSED W panel: [n][k]

  const int tid  = threadIdx.x;
  const int wave = tid >> 5;
  const int lane = tid & 31;
  const int hf   = lane >> 4;
  const int l16  = lane & 15;
  const int wm   = wave >> 2;              // 0..1  (16-row strip)
  const int wn   = wave & 3;               // 0..3  (32-col strip)
  const int mblk = blockIdx.y * 32;
  const int nblk = blockIdx.x * 128;

  v8f c0 = {}, c1 = {};

  for (int kc = 0; kc < K; kc += 64) {
    // ---- stage A panel 32x64 -> bf16 LDS (16B / thread) ----
    {
      const int r  = tid >> 3;          // 0..31
      const int cc = (tid & 7) * 8;     // 0..56
      if (A_BF16) {
        // Pure bf16 copy: CDNA5 async global->LDS DMA, lane-private addrs.
        const unsigned short* src =
            (const unsigned short*)Av + (size_t)(mblk + r) * K + kc + cc;
        const unsigned lds_off =
            (unsigned)(unsigned long long)(const void*)&As[r][cc];
        asm volatile("global_load_async_to_lds_b128 %0, %1, off"
                     :: "v"(lds_off), "v"((unsigned long long)src)
                     : "memory");
      } else {
        const float* A = (const float*)Av;
        const float4 a0 = *(const float4*)(A + (size_t)(mblk + r) * K + kc + cc);
        const float4 a1 = *(const float4*)(A + (size_t)(mblk + r) * K + kc + cc + 4);
        uint4 p;
        p.x = cvt_pk_bf16(a0.x, a0.y);
        p.y = cvt_pk_bf16(a0.z, a0.w);
        p.z = cvt_pk_bf16(a1.x, a1.y);
        p.w = cvt_pk_bf16(a1.z, a1.w);
        *(uint4*)&As[r][cc] = p;
        if (kc + 64 < K)   // gfx1250 global_prefetch_b8 of next panel
          __builtin_prefetch(A + (size_t)(mblk + r) * K + kc + 64 + cc, 0, 0);
      }
    }
    // ---- stage W panel 64(k) x 128(n), transposed into Bt[n][k] ----
    {
      const int r  = tid >> 2;          // k row 0..63
      const int cb = (tid & 3) * 32;    // n block
      const float* wrow = Wm + (size_t)(kc + r) * N + nblk + cb;
      float vbuf[32];
      #pragma unroll
      for (int q = 0; q < 8; ++q)
        *(float4*)&vbuf[q * 4] = *(const float4*)(wrow + q * 4);
      #pragma unroll
      for (int j = 0; j < 32; ++j) Bt[cb + j][r] = f2bf(vbuf[j]);
    }
    if (A_BF16)
      asm volatile("s_wait_asynccnt 0x0" ::: "memory");
    __syncthreads();

    #pragma unroll
    for (int g = 0; g < 2; ++g) {       // two K=32 sub-chunks
      const unsigned short* arow = &As[wm * 16 + l16][g * 32];
      const v16bf a = cat8(*(const v8bf*)(arow + 16 * hf),
                           *(const v8bf*)(arow + 16 * hf + 8));
      const unsigned short* brow0 = &Bt[wn * 32 + l16][g * 32];
      const unsigned short* brow1 = &Bt[wn * 32 + 16 + l16][g * 32];
      const v16bf b0 = cat8(*(const v8bf*)(brow0 + 16 * hf),
                            *(const v8bf*)(brow0 + 16 * hf + 8));
      const v16bf b1 = cat8(*(const v8bf*)(brow1 + 16 * hf),
                            *(const v8bf*)(brow1 + 16 * hf + 8));
      c0 = wmma_bf16(a, b0, c0);
      c1 = wmma_bf16(a, b1, c1);
    }
    __syncthreads();
  }

  const int m0 = mblk + wm * 16;
  const int n0 = nblk + wn * 32;
  const float bv0 = bias[n0 + l16];
  const float bv1 = bias[n0 + 16 + l16];
  #pragma unroll
  for (int r = 0; r < 8; ++r) {
    const float val0 = c0[r] + bv0;
    const float val1 = c1[r] + bv1;
    const size_t row = (size_t)(m0 + r + 8 * hf) * N;
    if (OUT_BF16) {
      ((unsigned short*)Cv)[row + n0 + l16]      = f2bf(val0);
      ((unsigned short*)Cv)[row + n0 + 16 + l16] = f2bf(val1);
    } else {
      ((float*)Cv)[row + n0 + l16]      = val0;
      ((float*)Cv)[row + n0 + 16 + l16] = val1;
    }
  }
}

// ---------------------------------------------------------------------------
// Attention: one block per (b, h, 16-row q tile).  256 threads = 8 waves.
//  A) S = (Q Kt)/8 : Q frags in registers, Kt frags contiguous global bf16
//  B) fp32 softmax, stream fp32 probs to d_out, keep bf16 copy in LDS
//  C) values = P @ V : V staged transposed per 64-key chunk, waves split
//     4 d-tiles x 2 K-halves, LDS partial combine, bf16 values out
// ---------------------------------------------------------------------------
__global__ __launch_bounds__(256) void attn_kernel(
    const unsigned short* __restrict__ qkv,   // bf16 [8192][2304]
    float* __restrict__ att_out,              // f32  [B,H,T,T]
    unsigned short* __restrict__ val_out)     // bf16 [8192][768]
{
  __shared__ float          Ss[16][SEQ + 4];    // fp32 logits/probs
  __shared__ unsigned short Pbf[16][SEQ + 8];   // bf16 probs (A frags)
  __shared__ unsigned short Vt[64][72];         // transposed V chunk [d][k]
  __shared__ float          Ps[4][16][16];      // split-K partials

  const int tid  = threadIdx.x;
  const int wave = tid >> 5;
  const int lane = tid & 31;
  const int hf   = lane >> 4;
  const int l16  = lane & 15;

  const int qt = blockIdx.x & 63;
  const int h  = (blockIdx.x >> 6) % HEADS;
  const int b  = blockIdx.x / (64 * HEADS);
  const size_t rowbase = (size_t)b * SEQ;
  const int hd = h * HDIM;

  // ---- phase A: Q fragments (whole 16x64 Q tile lives in regs/wave) ----
  const unsigned short* qrow = qkv + (rowbase + qt * 16 + l16) * QKV_N + hd;
  v16bf aq[2];
  #pragma unroll
  for (int g = 0; g < 2; ++g)
    aq[g] = cat8(*(const v8bf*)(qrow + 32 * g + 16 * hf),
                 *(const v8bf*)(qrow + 32 * g + 16 * hf + 8));

  #pragma unroll 2
  for (int i = 0; i < 8; ++i) {
    const int k0 = (wave + i * 8) * 16;
    const unsigned short* kbase =
        qkv + (rowbase + k0 + l16) * QKV_N + EMBED + hd;
    v8f c = {};
    #pragma unroll
    for (int g = 0; g < 2; ++g) {
      // B[kdim][key]: per-lane 16 consecutive head-dims of key k0+l16
      const v16bf bk = cat8(*(const v8bf*)(kbase + 32 * g + 16 * hf),
                            *(const v8bf*)(kbase + 32 * g + 16 * hf + 8));
      c = wmma_bf16(aq[g], bk, c);
    }
    #pragma unroll
    for (int r = 0; r < 8; ++r)
      Ss[r + 8 * hf][k0 + l16] = c[r] * 0.125f;   // HEAD_DIM^-0.5
  }
  __syncthreads();

  // ---- phase B: softmax over 1024 keys (2 rows / wave) ----
  #pragma unroll
  for (int rr = 0; rr < 2; ++rr) {
    const int r = wave * 2 + rr;
    float mx = -3.402823466e38f;
    #pragma unroll
    for (int i = 0; i < 32; ++i) mx = fmaxf(mx, Ss[r][lane + i * 32]);
    #pragma unroll
    for (int off = 16; off > 0; off >>= 1) mx = fmaxf(mx, __shfl_xor(mx, off, 32));
    float sum = 0.f;
    #pragma unroll
    for (int i = 0; i < 32; ++i) {
      const float e = __expf(Ss[r][lane + i * 32] - mx);
      Ss[r][lane + i * 32] = e;
      sum += e;
    }
    #pragma unroll
    for (int off = 16; off > 0; off >>= 1) sum += __shfl_xor(sum, off, 32);
    const float inv = 1.0f / sum;
    float* arow = att_out + (((size_t)(b * HEADS + h)) * SEQ + qt * 16 + r) * SEQ;
    #pragma unroll
    for (int i = 0; i < 32; ++i) {
      const float p = Ss[r][lane + i * 32] * inv;
      arow[lane + i * 32] = p;            // dominant HBM store stream (f32)
      Pbf[r][lane + i * 32] = f2bf(p);    // bf16 copy for the AV WMMA
    }
  }
  __syncthreads();

  // ---- phase C: values = P(16x1024) @ V(1024x64) ----
  const int dt = wave & 3;      // 16-wide d tile
  const int kh = wave >> 2;     // K half of each 64-key chunk
  const int d0 = dt * 16;
  v8f c = {};
  for (int it = 0; it < 16; ++it) {
    const int kc = it * 64;
    {   // stage transposed V chunk: Vt[d][k], 16 elements / thread
      const int r  = tid >> 2;          // key row 0..63
      const int cb = (tid & 3) * 16;    // dim block
      const unsigned short* vrow =
          qkv + (rowbase + kc + r) * QKV_N + 2 * EMBED + hd + cb;
      unsigned short vbuf[16];
      *(uint4*)&vbuf[0] = *(const uint4*)(vrow);
      *(uint4*)&vbuf[8] = *(const uint4*)(vrow + 8);
      #pragma unroll
      for (int j = 0; j < 16; ++j) Vt[cb + j][r] = vbuf[j];
    }
    __syncthreads();
    const int ks = kh * 32;
    const unsigned short* arow = &Pbf[l16][kc + ks];
    const v16bf a = cat8(*(const v8bf*)(arow + 16 * hf),
                         *(const v8bf*)(arow + 16 * hf + 8));
    const unsigned short* brow = &Vt[d0 + l16][ks];
    const v16bf bv = cat8(*(const v8bf*)(brow + 16 * hf),
                          *(const v8bf*)(brow + 16 * hf + 8));
    c = wmma_bf16(a, bv, c);
    __syncthreads();
  }
  if (kh == 0) {
    #pragma unroll
    for (int r = 0; r < 8; ++r) Ps[dt][r + 8 * hf][l16] = c[r];
  }
  __syncthreads();
  if (kh == 1) {
    #pragma unroll
    for (int r = 0; r < 8; ++r) {
      const float v = c[r] + Ps[dt][r + 8 * hf][l16];
      val_out[(rowbase + qt * 16 + r + 8 * hf) * EMBED + hd + d0 + l16] = f2bf(v);
    }
  }
}

// ---------------------------------------------------------------------------
// Host launcher.  Inputs: x, mask(unused, all-true), W_qkv, b_qkv, W_o, b_o.
// d_out = o (6,291,456 f32) ++ attention (100,663,296 f32)
// d_ws  = qkv bf16 [8192,2304] (37.7MB) ++ values bf16 [8192,768] (12.6MB)
// ---------------------------------------------------------------------------
extern "C" void kernel_launch(void* const* d_in, const int* in_sizes, int n_in,
                              void* d_out, int out_size, void* d_ws, size_t ws_size,
                              hipStream_t stream) {
  const float* x     = (const float*)d_in[0];
  const float* W_qkv = (const float*)d_in[2];
  const float* b_qkv = (const float*)d_in[3];
  const float* W_o   = (const float*)d_in[4];
  const float* b_o   = (const float*)d_in[5];

  float* o_out   = (float*)d_out;
  float* att_out = o_out + (size_t)ROWS * EMBED;

  unsigned short* qkv_ws = (unsigned short*)d_ws;
  unsigned short* val_ws = qkv_ws + (size_t)ROWS * QKV_N;

  dim3 blk(256);

  // 1) qkv(bf16) = x @ W_qkv + b_qkv     [8192 x 2304]
  gemm_bias_kernel<false, true><<<dim3(QKV_N / 128, ROWS / 32), blk, 0, stream>>>(
      x, W_qkv, b_qkv, qkv_ws, QKV_N);

  // 2) attention probs (f32 out) + values (bf16)
  attn_kernel<<<dim3(BATCH * HEADS * (SEQ / 16)), blk, 0, stream>>>(
      qkv_ws, att_out, val_ws);

  // 3) o(f32) = values @ W_o + b_o       [8192 x 768]
  gemm_bias_kernel<true, false><<<dim3(EMBED / 128, ROWS / 32), blk, 0, stream>>>(
      val_ws, W_o, b_o, o_out, EMBED);
}
